// HMM_28501402976723
// MI455X (gfx1250) — matrix operand order repaired and checked by reading, not compiled, not run on hardware
//
#include <hip/hip_runtime.h>

typedef __attribute__((ext_vector_type(16))) _Float16 v16h;
typedef __attribute__((ext_vector_type(8)))  float    v8f;

#define S_LEN 512
#define BATCH 256
#define ZD    64

// ---------------------------------------------------------------------------
// Layout trick: state x lives in WMMA D-layout (lane-half = bit3 of z).
// In contiguous k order B's lane-half = bit4 of K.  We define K = swap34(z) and
// permute the *columns of the constant A operand* accordingly at preload,
// so converting D-state -> B operand is purely lane-local:
//     B[k][i] = (f16) x[2k + (i>>3)][i&7]
// ---------------------------------------------------------------------------

// Load one emission row e[t][b][:] into the D-layout fragment of this lane.
__device__ __forceinline__ void load_e(const int* __restrict__ inp,
                                       const float* __restrict__ emit,
                                       int t, int b, int hi, float e[4][8]) {
  int idx = inp[t * BATCH + b];
  const float* er = emit + (size_t)idx * (size_t)ZD;
#pragma unroll
  for (int mt = 0; mt < 4; ++mt) {
    int z0 = 16 * mt + 8 * hi;
    float4 u = *(const float4*)(er + z0);
    float4 v = *(const float4*)(er + z0 + 4);
    e[mt][0] = u.x; e[mt][1] = u.y; e[mt][2] = u.z; e[mt][3] = u.w;
    e[mt][4] = v.x; e[mt][5] = v.y; e[mt][6] = v.z; e[mt][7] = v.w;
  }
}

// D-layout f32 state -> two f16 B operands.  Lane-local thanks to the
// swap34 column permutation folded into A; pairs (2i,2i+1) are adjacent
// halves of a VGPR, so this lowers to 16 v_cvt_pk_f16_f32 and nothing else.
__device__ __forceinline__ void build_B(const float x[4][8],
                                        v16h& B0, v16h& B1) {
#pragma unroll
  for (int k = 0; k < 2; ++k) {
    v16h bb;
#pragma unroll
    for (int j = 0; j < 8; ++j) {
      bb[j]     = (_Float16)x[2 * k][j];
      bb[8 + j] = (_Float16)x[2 * k + 1][j];
    }
    if (k == 0) B0 = bb; else B1 = bb;
  }
}

// d[mt] (16x16 f32 tile, z rows 16mt..16mt+15, cols = batch) = A(64x64) * B
__device__ __forceinline__ void mm(const v16h A[4][2], const v16h& B0,
                                   const v16h& B1, v8f d[4]) {
  v8f zf = {};
#pragma unroll
  for (int mt = 0; mt < 4; ++mt) {
    v8f acc = __builtin_amdgcn_wmma_f32_16x16x32_f16(
        false, A[mt][0], false, B0, (short)0, zf, false, false);
    d[mt] = __builtin_amdgcn_wmma_f32_16x16x32_f16(
        false, A[mt][1], false, B1, (short)0, acc, false, false);
  }
}

// Store this lane's D-layout fragment to out[t][b][:] (coalesced b128s).
__device__ __forceinline__ void store_rows(float* __restrict__ r,
                                           const float x[4][8], int hi) {
#pragma unroll
  for (int mt = 0; mt < 4; ++mt) {
    int z0 = 16 * mt + 8 * hi;
    *(float4*)(r + z0)     = make_float4(x[mt][0], x[mt][1], x[mt][2], x[mt][3]);
    *(float4*)(r + z0 + 4) = make_float4(x[mt][4], x[mt][5], x[mt][6], x[mt][7]);
  }
}

// ---------------------------------------------------------------------------
// Kernel 1: 32 single-wave workgroups; blocks 0..15 = forward chains (batch
// tiles of 16), blocks 16..31 = backward chains.  Recursion is transposed so
// the constant matrix T (resp. T^T) is the preloaded WMMA A operand.
// ---------------------------------------------------------------------------
__global__ __launch_bounds__(32) void hmm_chain_kernel(
    const int* __restrict__ inp, const float* __restrict__ T,
    const float* __restrict__ pi, const float* __restrict__ emit,
    float* __restrict__ alpha, float* __restrict__ beta) {
  const int lane = threadIdx.x & 31;
  const int hi   = lane >> 4;
  const int c    = lane & 15;
  const int wid  = blockIdx.x;
  const bool fwd = wid < 16;
  const int b    = ((wid & 15) << 4) + c;   // global batch index for this lane

  // ---- Preload A (f16, A-layout) with swap34-permuted columns.
  // Hardware slot K = 32*kc + 16*g + 8*hi + jj  maps to logical column
  // kcol = 32*kc + 16*g + 8*hi + jj  (table order, matching B).
  v16h A[4][2];
#pragma unroll
  for (int mt = 0; mt < 4; ++mt)
#pragma unroll
    for (int kc = 0; kc < 2; ++kc)
#pragma unroll
      for (int g = 0; g < 2; ++g) {
        int kcol0 = 32 * kc + 8 * hi + 16 * g;
        int m     = 16 * mt + c;
        if (fwd) {
          const float* p = T + m * ZD + kcol0;      // A = T
#pragma unroll
          for (int jj = 0; jj < 8; ++jj)
            A[mt][kc][8 * g + jj] = (_Float16)p[jj];
        } else {
#pragma unroll
          for (int jj = 0; jj < 8; ++jj)            // A = T^T
            A[mt][kc][8 * g + jj] = (_Float16)T[(kcol0 + jj) * ZD + m];
        }
      }

  if (fwd) {
    // ---- forward:  x_t = e_t^T  ⊙ (T @ x_{t-1}),  x = alpha^T
    float x[4][8];
    {
      // t = 0 : x = e0 * pi (broadcast over batch)
      load_e(inp, emit, 0, b, hi, x);
#pragma unroll
      for (int mt = 0; mt < 4; ++mt) {
        int z0 = 16 * mt + 8 * hi;
        float4 pu = *(const float4*)(pi + z0);
        float4 pv = *(const float4*)(pi + z0 + 4);
        x[mt][0] *= pu.x; x[mt][1] *= pu.y; x[mt][2] *= pu.z; x[mt][3] *= pu.w;
        x[mt][4] *= pv.x; x[mt][5] *= pv.y; x[mt][6] *= pv.z; x[mt][7] *= pv.w;
      }
      store_rows(alpha + (size_t)b * ZD, x, hi);
    }
    for (int t = 1; t < S_LEN; ++t) {
      v16h B0, B1;
      build_B(x, B0, B1);
      v8f d[4];
      mm(A, B0, B1, d);
      load_e(inp, emit, t, b, hi, x);
#pragma unroll
      for (int mt = 0; mt < 4; ++mt)
#pragma unroll
        for (int j = 0; j < 8; ++j)
          x[mt][j] *= d[mt][j];
      store_rows(alpha + ((size_t)t * BATCH + b) * ZD, x, hi);
    }
  } else {
    // ---- backward: y_t = T^T @ (e_{t+1}^T ⊙ y_{t+1}),  y = beta^T
    float y[4][8];
#pragma unroll
    for (int mt = 0; mt < 4; ++mt)
#pragma unroll
      for (int j = 0; j < 8; ++j)
        y[mt][j] = 1.0f;
    store_rows(beta + ((size_t)(S_LEN - 1) * BATCH + b) * ZD, y, hi);

    for (int t = S_LEN - 2; t >= 0; --t) {
      float w[4][8];
      load_e(inp, emit, t + 1, b, hi, w);
#pragma unroll
      for (int mt = 0; mt < 4; ++mt)
#pragma unroll
        for (int j = 0; j < 8; ++j)
          w[mt][j] *= y[mt][j];
      v16h B0, B1;
      build_B(w, B0, B1);
      v8f d[4];
      mm(A, B0, B1, d);
#pragma unroll
      for (int mt = 0; mt < 4; ++mt)
#pragma unroll
        for (int j = 0; j < 8; ++j)
          y[mt][j] = d[mt][j];
      store_rows(beta + ((size_t)t * BATCH + b) * ZD, y, hi);
    }
  }
}

// ---------------------------------------------------------------------------
// Kernel 2: posterior = (alpha*beta) / sum_z(alpha*beta).  One wave per
// (t,b) row: float2 per lane covers the 64-wide Z row, wave shfl reduction.
// Pure bandwidth pass (~100 MB of traffic).
// ---------------------------------------------------------------------------
__global__ __launch_bounds__(128) void hmm_post_kernel(
    const float* __restrict__ alpha, const float* __restrict__ beta,
    float* __restrict__ post) {
  int lane   = threadIdx.x & 31;
  size_t row = (size_t)blockIdx.x * 4 + (threadIdx.x >> 5);
  const float2* ar = (const float2*)(alpha + row * ZD);
  const float2* br = (const float2*)(beta + row * ZD);
  float2 a2 = ar[lane];
  float2 b2 = br[lane];
  float p0 = a2.x * b2.x;
  float p1 = a2.y * b2.y;
  float s = p0 + p1;
#pragma unroll
  for (int off = 16; off >= 1; off >>= 1)
    s += __shfl_xor(s, off, 32);
  float inv = 1.0f / s;
  ((float2*)(post + row * ZD))[lane] = make_float2(p0 * inv, p1 * inv);
}

// ---------------------------------------------------------------------------
extern "C" void kernel_launch(void* const* d_in, const int* in_sizes, int n_in,
                              void* d_out, int out_size, void* d_ws,
                              size_t ws_size, hipStream_t stream) {
  (void)in_sizes; (void)n_in; (void)out_size; (void)d_ws; (void)ws_size;
  const int*   inp  = (const int*)d_in[0];
  const float* T    = (const float*)d_in[1];
  const float* pi   = (const float*)d_in[2];
  const float* emit = (const float*)d_in[3];

  float* alpha = (float*)d_out;
  float* beta  = alpha + (size_t)S_LEN * BATCH * ZD;
  float* post  = beta  + (size_t)S_LEN * BATCH * ZD;

  hipLaunchKernelGGL(hmm_chain_kernel, dim3(32), dim3(32), 0, stream,
                     inp, T, pi, emit, alpha, beta);
  hipLaunchKernelGGL(hmm_post_kernel, dim3((S_LEN * BATCH) / 4), dim3(128), 0,
                     stream, alpha, beta, post);
}
